// MultiheadAttention_82274393522942
// MI455X (gfx1250) — compile-verified, matrix operand order checked by reading, not run on hardware
//
#include <hip/hip_runtime.h>

// ---------------------------------------------------------------------------
// MultiheadAttention for MI455X (gfx1250): bf16 WMMA, fp32 accum,
// TDM (tensor_load_to_lds) weight staging, fused softmax attention.
// B=2, S=S'=2048, H=16, D_MODEL=1024, DK=DV=64.
// d_out = [ out fp32 (2*2048*1024) | attn fp32 (2*16*2048*2048) ]
// ---------------------------------------------------------------------------

#define BQ      2
#define SEQ     2048
#define DMODEL  1024
#define NHEAD   16
#define DK      64
#define ROWS    (BQ * SEQ)
#define ATTN_OFF ((size_t)ROWS * DMODEL)

typedef __attribute__((ext_vector_type(16))) __bf16 v16bf;
typedef __attribute__((ext_vector_type(8)))  float  v8f;
typedef unsigned int v4u __attribute__((ext_vector_type(4)));
typedef int          v8i __attribute__((ext_vector_type(8)));
typedef int          v4i __attribute__((ext_vector_type(4)));

static __device__ __forceinline__ __bf16 f2bf(float x) {
  union { float f; unsigned u; } v; v.f = x;
  unsigned r = v.u + 0x7FFFu + ((v.u >> 16) & 1u);   // round-to-nearest-even
  unsigned short h = (unsigned short)(r >> 16);
  return __builtin_bit_cast(__bf16, h);
}

// 16x16x32 bf16 WMMA fragment from row-major K-contiguous storage.
// Lane L: row = L&15, K-half = (L>>4)*16 -> one 32B contiguous load per lane.
static __device__ __forceinline__ v16bf load_frag(const __bf16* base, int ld, int lane) {
  int row = lane & 15;
  int kb  = (lane >> 4) << 4;
  return *reinterpret_cast<const v16bf*>(base + (size_t)row * ld + kb);
}

static __device__ __forceinline__ v8f wmma_bf16(v16bf a, v16bf b, v8f c) {
  return __builtin_amdgcn_wmma_f32_16x16x32_bf16(false, a, false, b, (short)0, c,
                                                 false, false);
}

// C/D tile (16x16 f32): lane L col = L&15, rows (L>>4)*8 + i.
static __device__ __forceinline__ void store_tile_bf16(__bf16* C, int ldc, int lane, v8f acc) {
  int col = lane & 15;
  int rb  = (lane >> 4) << 3;
#pragma unroll
  for (int i = 0; i < 8; ++i) C[(size_t)(rb + i) * ldc + col] = f2bf(acc[i]);
}

// TDM: stage tile [rows x K] bf16 (row stride K) from `g` to LDS offset 0.
// D# per CDNA5 ISA; 6-arg amdgpu-toolchain builtin form.
static __device__ __forceinline__ void tdm_stage_to_lds(const __bf16* g, int K,
                                                        int rows, int rows_avail) {
  unsigned long long ga = (unsigned long long)(uintptr_t)g;
  v4u g0;
  g0[0] = 1u;                                   // count=1 valid descriptor
  g0[1] = 0u;                                   // lds_addr = 0 (dynamic LDS base)
  g0[2] = (unsigned)ga;                         // global_addr[31:0]
  g0[3] = (unsigned)((ga >> 32) & 0x01FFFFFFu) | 0x80000000u;  // addr[56:32] | type=2
  v8i g1;
  g1[0] = 0x00010000;                           // wg_mask=0, data_size=1 (2 bytes)
  g1[1] = (K & 0xFFFF) << 16;                   // tensor_dim0[15:0]
  g1[2] = ((rows_avail & 0xFFFF) << 16) | ((K >> 16) & 0xFFFF);   // dim1lo|dim0hi
  g1[3] = ((K & 0xFFFF) << 16) | ((rows_avail >> 16) & 0xFFFF);   // tile_dim0|dim1hi
  g1[4] = rows & 0xFFFF;                        // tile_dim1
  g1[5] = K;                                    // tensor_dim0_stride[31:0]
  g1[6] = 0;
  g1[7] = 0;
  v4i z4 = {0, 0, 0, 0};
  v8i z8 = {0, 0, 0, 0, 0, 0, 0, 0};
  __builtin_amdgcn_tensor_load_to_lds(g0, g1, z4, z4, z8, 0);
  __builtin_amdgcn_s_wait_tensorcnt(0);
}

// ---------------------------------------------------------------------------
// LayerNorm(query) -> bf16, one block per row of 1024
// ---------------------------------------------------------------------------
__global__ __launch_bounds__(256)
void ln_to_bf16(const float* __restrict__ x, const float* __restrict__ g,
                const float* __restrict__ b, __bf16* __restrict__ out) {
  __shared__ float rs[256], rs2[256];
  int row = blockIdx.x, t = threadIdx.x;
  const float* xr = x + (size_t)row * DMODEL;
  float v[4], s = 0.f, s2 = 0.f;
#pragma unroll
  for (int i = 0; i < 4; ++i) { v[i] = xr[t + i * 256]; s += v[i]; s2 += v[i] * v[i]; }
  rs[t] = s; rs2[t] = s2; __syncthreads();
  for (int off = 128; off > 0; off >>= 1) {
    if (t < off) { rs[t] += rs[t + off]; rs2[t] += rs2[t + off]; }
    __syncthreads();
  }
  float mean = rs[0] * (1.0f / DMODEL);
  float var  = rs2[0] * (1.0f / DMODEL) - mean * mean;
  float rinv = rsqrtf(var + 1e-6f);
#pragma unroll
  for (int i = 0; i < 4; ++i) {
    int c = t + i * 256;
    out[(size_t)row * DMODEL + c] = f2bf((v[i] - mean) * rinv * g[c] + b[c]);
  }
}

__global__ __launch_bounds__(256)
void cast_bf16(const float* __restrict__ in, __bf16* __restrict__ out, int n) {
  for (int i = blockIdx.x * blockDim.x + threadIdx.x; i < n; i += gridDim.x * blockDim.x)
    out[i] = f2bf(in[i]);
}

// ---------------------------------------------------------------------------
// C[4096,1024] = A[4096,1024] @ W[1024,1024]^T (all dims compile-time).
// Workgroup: 8 waves, 128(M) x 64(N) tile. W block-row (64x1024, 128 KB)
// staged to LDS via TDM. Inner loop: A ping-pong (a0/a1) prefetched from
// global; per half-step all 4 B-frags batched from LDS (8 ds_loads ->
// one wait -> 4 back-to-back WMMAs). Grid (1024/64, 4096/128).
// ---------------------------------------------------------------------------
#define GK DMODEL

static __device__ __forceinline__ void gemm_core(const __bf16* __restrict__ A,
                                                 const __bf16* wt, int m0,
                                                 int lane, v8f acc[4]) {
  const __bf16* arow = A + (size_t)m0 * GK;
  int kb = (lane >> 4) << 4, nl = lane & 15;
  v16bf a0 = load_frag(arow, GK, lane);
  v16bf a1 = load_frag(arow + 32, GK, lane);
  for (int k0 = 0; k0 < GK; k0 += 64) {
    v16bf an0 = a0, an1 = a1;
    if (k0 + 64 < GK) {                          // global prefetch next pair
      an0 = load_frag(arow + k0 + 64, GK, lane);
      an1 = load_frag(arow + k0 + 96, GK, lane);
    }
    v16bf bf[4];
#pragma unroll
    for (int j = 0; j < 4; ++j)
      bf[j] = *reinterpret_cast<const v16bf*>(wt + (size_t)(j * 16 + nl) * GK + k0 + kb);
#pragma unroll
    for (int j = 0; j < 4; ++j) acc[j] = wmma_bf16(a0, bf[j], acc[j]);
#pragma unroll
    for (int j = 0; j < 4; ++j)
      bf[j] = *reinterpret_cast<const v16bf*>(wt + (size_t)(j * 16 + nl) * GK + k0 + 32 + kb);
#pragma unroll
    for (int j = 0; j < 4; ++j) acc[j] = wmma_bf16(a1, bf[j], acc[j]);
    a0 = an0; a1 = an1;
  }
}

__global__ __launch_bounds__(256)
void gemm_bf16_nt(const __bf16* __restrict__ A, const __bf16* __restrict__ W,
                  __bf16* __restrict__ C) {
  extern __shared__ __bf16 wt[];                 // [64][1024]
  int wave = threadIdx.x >> 5, lane = threadIdx.x & 31;
  int m0 = blockIdx.y * 128 + wave * 16;
  int nb = blockIdx.x * 64;
  if (wave == 0)
    tdm_stage_to_lds(W + (size_t)nb * GK, GK, 64, DMODEL - nb);
  __syncthreads();

  v8f acc[4] = {};
  gemm_core(A, wt, m0, lane, acc);
#pragma unroll
  for (int j = 0; j < 4; ++j)
    store_tile_bf16(C + (size_t)m0 * DMODEL + nb + j * 16, DMODEL, lane, acc[j]);
}

// Same GEMM, fp32 output + residual (final output projection).
__global__ __launch_bounds__(256)
void gemm_bf16_nt_res_f32(const __bf16* __restrict__ A, const __bf16* __restrict__ W,
                          const float* __restrict__ res, float* __restrict__ C) {
  extern __shared__ __bf16 wt[];
  int wave = threadIdx.x >> 5, lane = threadIdx.x & 31;
  int m0 = blockIdx.y * 128 + wave * 16;
  int nb = blockIdx.x * 64;
  if (wave == 0)
    tdm_stage_to_lds(W + (size_t)nb * GK, GK, 64, DMODEL - nb);
  __syncthreads();

  v8f acc[4] = {};
  gemm_core(A, wt, m0, lane, acc);
  int nl = lane & 15, rb = (lane >> 4) << 3;
#pragma unroll
  for (int j = 0; j < 4; ++j) {
    int n = nb + j * 16 + nl;
#pragma unroll
    for (int i = 0; i < 8; ++i) {
      size_t idx = (size_t)(m0 + rb + i) * DMODEL + n;
      C[idx] = acc[j][i] + res[idx];
    }
  }
}

// ---------------------------------------------------------------------------
// Per-head V transpose: vT[b][h][d][t] = vproj[b*S'+t][h*64+d]  (bf16)
// ---------------------------------------------------------------------------
__global__ __launch_bounds__(256)
void transpose_v(const __bf16* __restrict__ vp, __bf16* __restrict__ vT) {
  int total = BQ * NHEAD * DK * SEQ;
  for (int idx = blockIdx.x * blockDim.x + threadIdx.x; idx < total;
       idx += gridDim.x * blockDim.x) {
    int t = idx & (SEQ - 1);
    int d = (idx >> 11) & (DK - 1);
    int h = (idx >> 17) & (NHEAD - 1);
    int b = idx >> 21;
    vT[idx] = vp[((size_t)(b * SEQ + t)) * DMODEL + h * DK + d];
  }
}

// ---------------------------------------------------------------------------
// Fused attention: one block = one (b, h, 16-row s-tile).
//   P1: scores(16x2048) = Q K^T / 8 via WMMA -> LDS (Q frags hoisted,
//       K frags double-buffered)
//   P2: softmax in LDS
//   P3: normalize + stream attn fp32 to HBM (float4 / b128 stores — the
//       HBM-bound leg: 537 MB of attention probabilities)
//   P4: out(16x64) = attn @ V via WMMA, A from LDS (float4 + cvt), B frags
//       contiguous from vT (double-buffered)
// Dynamic LDS: 16*2048 f32 + 256 + 16 = 132,160 B (<320 KB/WGP).
// ---------------------------------------------------------------------------
__global__ __launch_bounds__(256)
void attn_fused(const __bf16* __restrict__ qp, const __bf16* __restrict__ kp,
                const __bf16* __restrict__ vT, float* __restrict__ attn,
                __bf16* __restrict__ ows) {
  extern __shared__ float smem[];
  float* sc      = smem;                 // [16][2048]
  float* red     = smem + 16 * SEQ;      // [256]
  float* rowstat = red + 256;            // [16]

  int t    = threadIdx.x;
  int wave = t >> 5, lane = t & 31;
  int st = blockIdx.x & 127;
  int h  = (blockIdx.x >> 7) & 15;
  int b  = blockIdx.x >> 11;
  int s0 = st * 16;

  const __bf16* qbase = qp + ((size_t)(b * SEQ + s0)) * DMODEL + h * DK;
  const __bf16* kbase = kp + ((size_t)b * SEQ) * DMODEL + h * DK;
  const __bf16* vbase = vT + ((size_t)(b * NHEAD + h) * DK) * SEQ;

  // ---- Phase 1: scores -> LDS (Q frags invariant; K frags pipelined) ----
  {
    v16bf aq0 = load_frag(qbase + 0,  DMODEL, lane);
    v16bf aq1 = load_frag(qbase + 32, DMODEL, lane);
    v16bf kb0 = load_frag(kbase + (size_t)wave * 16 * DMODEL + 0,  DMODEL, lane);
    v16bf kb1 = load_frag(kbase + (size_t)wave * 16 * DMODEL + 32, DMODEL, lane);
    int col = lane & 15, rb = (lane >> 4) << 3;
    for (int tt = wave; tt < SEQ / 16; tt += 8) {
      v16bf kn0 = kb0, kn1 = kb1;
      int tn = tt + 8;
      if (tn < SEQ / 16) {
        kn0 = load_frag(kbase + (size_t)tn * 16 * DMODEL + 0,  DMODEL, lane);
        kn1 = load_frag(kbase + (size_t)tn * 16 * DMODEL + 32, DMODEL, lane);
      }
      v8f acc = {};
      acc = wmma_bf16(aq0, kb0, acc);
      acc = wmma_bf16(aq1, kb1, acc);
#pragma unroll
      for (int i = 0; i < 8; ++i) sc[(rb + i) * SEQ + tt * 16 + col] = acc[i] * 0.125f;
      kb0 = kn0; kb1 = kn1;
    }
  }
  __syncthreads();

  // ---- Phase 2: softmax (16 threads per row, 128 cols each) ----
  {
    int r = t >> 4, seg = t & 15;
    float m = -3.0e38f;
    for (int c = seg * 128; c < seg * 128 + 128; ++c) m = fmaxf(m, sc[r * SEQ + c]);
    red[t] = m; __syncthreads();
    if (seg == 0) {
      float mm = red[t];
#pragma unroll
      for (int i = 1; i < 16; ++i) mm = fmaxf(mm, red[t + i]);
      rowstat[r] = mm;
    }
    __syncthreads();
    float rm = rowstat[r], sum = 0.f;
    for (int c = seg * 128; c < seg * 128 + 128; ++c) {
      float e = __expf(sc[r * SEQ + c] - rm);
      sc[r * SEQ + c] = e; sum += e;
    }
    red[t] = sum; __syncthreads();
    if (seg == 0) {
      float ss = red[t];
#pragma unroll
      for (int i = 1; i < 16; ++i) ss += red[t + i];
      rowstat[r] = 1.0f / ss;
    }
    __syncthreads();
  }

  // ---- Phase 3: normalize + stream attn (float4 -> global_store_b128) ----
  {
    float4* abase4 = (float4*)(attn + (((size_t)(b * NHEAD + h) * SEQ + s0) * SEQ));
    for (int idx = t; idx < 16 * (SEQ / 4); idx += 256) {
      int r = idx >> 9, c4 = idx & (SEQ / 4 - 1);
      float4* sp = (float4*)(sc + r * SEQ) + c4;
      float4 vv = *sp;
      float inv = rowstat[r];
      vv.x *= inv; vv.y *= inv; vv.z *= inv; vv.w *= inv;
      *sp = vv;
      abase4[(size_t)r * (SEQ / 4) + c4] = vv;
    }
  }
  __syncthreads();

  // ---- Phase 4: out = attn @ V (waves 0..3, one 16x16 d-tile each) ----
  if (wave < 4) {
    int d0 = wave * 16;
    int row = lane & 15, kb = (lane >> 4) << 4;
    const __bf16* vrow = vbase + (size_t)(d0 + (lane & 15)) * SEQ;
    v16bf bcur = *reinterpret_cast<const v16bf*>(vrow + kb);
    v8f acc = {};
    for (int k0 = 0; k0 < SEQ; k0 += 32) {
      v16bf bnext = bcur;
      if (k0 + 32 < SEQ)
        bnext = *reinterpret_cast<const v16bf*>(vrow + k0 + 32 + kb);
      v16bf a;
      const float4* sp = (const float4*)(sc + row * SEQ + k0 + kb);
#pragma unroll
      for (int q = 0; q < 4; ++q) {
        float4 f = sp[q];
        a[4 * q + 0] = f2bf(f.x); a[4 * q + 1] = f2bf(f.y);
        a[4 * q + 2] = f2bf(f.z); a[4 * q + 3] = f2bf(f.w);
      }
      acc = wmma_bf16(a, bcur, acc);
      bcur = bnext;
    }
    store_tile_bf16(ows + ((size_t)(b * SEQ + s0)) * DMODEL + h * DK + d0,
                    DMODEL, lane, acc);
  }
}

// ---------------------------------------------------------------------------
extern "C" void kernel_launch(void* const* d_in, const int* in_sizes, int n_in,
                              void* d_out, int out_size, void* d_ws, size_t ws_size,
                              hipStream_t stream) {
  const float* query = (const float*)d_in[0];
  const float* key   = (const float*)d_in[1];
  const float* value = (const float*)d_in[2];
  const float* w_q   = (const float*)d_in[3];
  const float* w_k   = (const float*)d_in[4];
  const float* w_v   = (const float*)d_in[5];
  const float* w_o   = (const float*)d_in[6];
  const float* ln_g  = (const float*)d_in[7];
  const float* ln_b  = (const float*)d_in[8];
  float* out = (float*)d_out;

  // Workspace layout (bf16 tensors), ~72 MiB total.
  char* ws = (char*)d_ws;
  const size_t ACT = (size_t)ROWS * DMODEL * sizeof(__bf16);   // 8 MiB
  const size_t WSZ = (size_t)DMODEL * DMODEL * sizeof(__bf16); // 2 MiB
  __bf16* qn   = (__bf16*)(ws);
  __bf16* keyb = (__bf16*)(ws + ACT);
  __bf16* valb = (__bf16*)(ws + 2 * ACT);
  __bf16* wqb  = (__bf16*)(ws + 3 * ACT);
  __bf16* wkb  = (__bf16*)(ws + 3 * ACT + WSZ);
  __bf16* wvb  = (__bf16*)(ws + 3 * ACT + 2 * WSZ);
  __bf16* wob  = (__bf16*)(ws + 3 * ACT + 3 * WSZ);
  __bf16* qpj  = (__bf16*)(ws + 3 * ACT + 4 * WSZ);
  __bf16* kpj  = (__bf16*)(ws + 4 * ACT + 4 * WSZ);
  __bf16* vpj  = (__bf16*)(ws + 5 * ACT + 4 * WSZ);
  __bf16* vT   = (__bf16*)(ws + 6 * ACT + 4 * WSZ);
  __bf16* ows  = (__bf16*)(ws + 7 * ACT + 4 * WSZ);

  const int nAct = ROWS * DMODEL;
  const int nW   = DMODEL * DMODEL;

  ln_to_bf16<<<ROWS, 256, 0, stream>>>(query, ln_g, ln_b, qn);
  cast_bf16<<<2048, 256, 0, stream>>>(key,   keyb, nAct);
  cast_bf16<<<2048, 256, 0, stream>>>(value, valb, nAct);
  cast_bf16<<<1024, 256, 0, stream>>>(w_q, wqb, nW);
  cast_bf16<<<1024, 256, 0, stream>>>(w_k, wkb, nW);
  cast_bf16<<<1024, 256, 0, stream>>>(w_v, wvb, nW);
  cast_bf16<<<1024, 256, 0, stream>>>(w_o, wob, nW);

  // Q/K/V projections: [4096,1024] @ [1024,1024]^T, TDM-staged weights
  dim3 gg(DMODEL / 64, ROWS / 128);
  size_t wlds = (size_t)64 * DMODEL * sizeof(__bf16);  // 128 KiB
  gemm_bf16_nt<<<gg, 256, wlds, stream>>>(qn,   wqb, qpj);
  gemm_bf16_nt<<<gg, 256, wlds, stream>>>(keyb, wkb, kpj);
  gemm_bf16_nt<<<gg, 256, wlds, stream>>>(valb, wvb, vpj);

  transpose_v<<<4096, 256, 0, stream>>>(vpj, vT);

  size_t alds = (16 * SEQ + 256 + 16) * sizeof(float);  // 132,160 B
  attn_fused<<<BQ * NHEAD * (SEQ / 16), 256, alds, stream>>>(
      qpj, kpj, vT, out + ATTN_OFF, ows);

  gemm_bf16_nt_res_f32<<<gg, 256, wlds, stream>>>(ows, wob, query, out);
}
